// SelfAttention_2_27547920237135
// MI455X (gfx1250) — compile-verified, matrix operand order checked by reading, not run on hardware
//
#include <hip/hip_runtime.h>
#include <hip/hip_bf16.h>
#include <stdint.h>

// Self-attention, fused flash-attention for gfx1250 (wave32, WMMA bf16).
// B=2, C=512, Cqk=64, N=4096.  Manual loadcnt-pipelined inner loops with
// token-injected waits (s_wait_loadcnt produces an opaque 0 that is OR'd
// into each guarded operand -> consuming WMMAs cannot hoist above the wait).

typedef __attribute__((ext_vector_type(16))) __bf16 v16bf;
typedef __attribute__((ext_vector_type(8)))  float  v8f;

union Op16 {
  v16bf v;
  __bf16 h[16];
  unsigned short e[16];
  uint4 u[2];
};

// ---- inline-asm primitives ------------------------------------------------
#define GLB128(dst, ptr, OFF)                                   \
  asm volatile("global_load_b128 %0, %1, off offset:" OFF       \
               : "=v"(dst) : "v"(ptr))

#define WAIT_TOK(N, tokvar)                                     \
  asm volatile("s_wait_loadcnt " N "\n\ts_mov_b32 %0, 0"        \
               : "=s"(tokvar))

#define TOK_ONLY(tokvar)                                        \
  asm volatile("s_mov_b32 %0, 0" : "=s"(tokvar))

// ---------------------------------------------------------------------------
// Prep kernel A: x [B,C,N] f32  ->  Xt [B,N,C] bf16 (LDS-tiled transpose).
// ---------------------------------------------------------------------------
__global__ __launch_bounds__(256)
void xpose_kernel(const float* __restrict__ x, __bf16* __restrict__ Xt)
{
  __shared__ __bf16 lds[64 * 72];

  const int blk = blockIdx.x;
  const int b   = blk >> 9;
  const int r   = blk & 511;
  const int c0  = (r >> 6) * 64;
  const int n0  = (r & 63) * 64;
  const float* xb = x + (size_t)b * 512 * 4096;

  const int t      = threadIdx.x;
  const int nlocal = t & 63;
  const int cgrp   = t >> 6;
#pragma unroll
  for (int i = 0; i < 16; ++i) {
    int c = cgrp * 16 + i;
    float f = xb[(size_t)(c0 + c) * 4096 + n0 + nlocal];
    lds[nlocal * 72 + c] = (__bf16)f;
  }
  __syncthreads();

  const int n  = t >> 2;
  const int cs = (t & 3) * 16;
  const uint4* src = (const uint4*)&lds[n * 72 + cs];
  uint4* dst = (uint4*)(Xt + ((size_t)b * 4096 + n0 + n) * 512 + c0 + cs);
  dst[0] = src[0];
  dst[1] = src[1];
}

// ---------------------------------------------------------------------------
// Prep kernel B: flat f32 -> bf16 convert (weights).
// ---------------------------------------------------------------------------
__global__ __launch_bounds__(256)
void cvt_kernel(const float* __restrict__ src, __bf16* __restrict__ dst, int n)
{
  int i = blockIdx.x * 256 + threadIdx.x;
  if (i < n) dst[i] = (__bf16)src[i];
}

// ---------------------------------------------------------------------------
// Kernel 1: QKV projection GEMMs, manually modulo-scheduled (2 groups deep).
// ---------------------------------------------------------------------------
#define K1_LOAD_GROUP(BUF, XADDR, WADDR)        \
  GLB128(bm[BUF].u[0], (XADDR), "0");           \
  GLB128(bm[BUF].u[1], (XADDR), "32");          \
  GLB128(am[BUF][0].u[0], (WADDR), "0");        \
  GLB128(am[BUF][0].u[1], (WADDR), "32");       \
  GLB128(am[BUF][1].u[0], (WADDR), "16384");    \
  GLB128(am[BUF][1].u[1], (WADDR), "16416");    \
  GLB128(am[BUF][2].u[0], (WADDR), "32768");    \
  GLB128(am[BUF][2].u[1], (WADDR), "32800");    \
  GLB128(am[BUF][3].u[0], (WADDR), "49152");    \
  GLB128(am[BUF][3].u[1], (WADDR), "49184");

#define K1_WAIT(BUF, CNT) do { unsigned _t;                     \
  WAIT_TOK(CNT, _t);                                            \
  bm[BUF].u[0].x    |= _t;                                      \
  am[BUF][0].u[0].x |= _t; am[BUF][1].u[0].x |= _t;             \
  am[BUF][2].u[0].x |= _t; am[BUF][3].u[0].x |= _t;             \
  tokf = _t; } while (0)

#define K1_WMMA(BUF)                                                           \
  _Pragma("unroll")                                                            \
  for (int m = 0; m < 4; ++m) {                                                \
    acc[m] = __builtin_amdgcn_wmma_f32_16x16x32_bf16(                          \
        false, am[BUF][m].v, false, bm[BUF].v, (short)0, acc[m], false, false);\
  }

__global__ __launch_bounds__(128)
void qkv_project_kernel(const __bf16* __restrict__ Xt,
                        const __bf16* __restrict__ wqb, const float* __restrict__ bq,
                        const __bf16* __restrict__ wkb, const float* __restrict__ bk,
                        const __bf16* __restrict__ wvb, const float* __restrict__ bv,
                        __bf16* __restrict__ Q,
                        __bf16* __restrict__ K,
                        __bf16* __restrict__ V)
{
  const int lane = threadIdx.x & 31;
  const int wave = threadIdx.x >> 5;
  const int half = lane >> 4;
  const int l15  = lane & 15;

  int tid = blockIdx.x * 4 + wave;
  int b   = tid / (10 * 256);
  int rem = tid % (10 * 256);
  int mg  = rem / 256;                       // 0=Q, 1=K, 2..9=V d-group
  int n0  = (rem % 256) * 16;

  const __bf16* W; const float* bias; int dbase;
  __bf16* out; bool transposed;
  if (mg == 0)      { W = wqb; bias = bq; dbase = 0;          out = Q + (size_t)b*4096*64;  transposed = true;  }
  else if (mg == 1) { W = wkb; bias = bk; dbase = 0;          out = K + (size_t)b*4096*64;  transposed = true;  }
  else              { W = wvb; bias = bv; dbase = (mg-2)*64;  out = V + (size_t)b*512*4096; transposed = false; }

  const char* xaddr = (const char*)(Xt + ((size_t)b * 4096 + n0 + l15) * 512 + half * 8);
  const char* waddr = (const char*)(W + (size_t)(dbase + l15) * 512 + half * 8);

  Op16 bm[2];
  Op16 am[2][4];
  v8f acc[4] = {};
  unsigned tokf = 0;

  K1_LOAD_GROUP(0, xaddr, waddr);
  K1_LOAD_GROUP(1, xaddr + 64, waddr + 64);

#pragma unroll
  for (int step = 0; step < 14; ++step) {
    K1_WAIT(step & 1, "0xa");                // current group landed; next in flight
    K1_WMMA(step & 1);
    int cn = (step + 2) * 64;                // byte offset of step+2 (no wrap needed)
    K1_LOAD_GROUP(step & 1, xaddr + cn, waddr + cn);
  }
  // peeled tail: steps 14 and 15, no further issue, final wait drains all
  K1_WAIT(0, "0xa");
  K1_WMMA(0);
  K1_WAIT(1, "0x0");                         // loadcnt==0 after this; tokf set
  K1_WMMA(1);

  // epilogue (token-poisoned so nothing hoists above the final drain)
  n0    += (int)tokf;
  dbase += (int)tokf;
  bias  += tokf;
  out   += tokf;
#pragma unroll
  for (int m = 0; m < 4; ++m) {
#pragma unroll
    for (int v = 0; v < 8; ++v) {
      int d = dbase + m*16 + v + half*8;
      int n = n0 + l15;
      float val = acc[m][v] + bias[d];
      if (transposed) out[(size_t)n * 64   + d] = (__bf16)val;
      else            out[(size_t)d * 4096 + n] = (__bf16)val;
    }
  }
}

// ---------------------------------------------------------------------------
// Kernel 2: fused flash attention + epilogue (gamma*out + x).
// Block = 4 waves; wave w owns channels [w*128,+128).
// Pipeline per 32-kv step:
//   wait(K, tok->kb) -> S wmma x4 -> issue V(16) [+ nextK(8)] -> softmax/LDS
//   -> accL wmma -> wait(V, tok->vb) -> PV wmma x8
// ---------------------------------------------------------------------------
#define K2_LOAD_K(KADDR)                    \
  GLB128(kb[0].u[0], (KADDR), "0");         \
  GLB128(kb[0].u[1], (KADDR), "32");        \
  GLB128(kb[1].u[0], (KADDR), "64");        \
  GLB128(kb[1].u[1], (KADDR), "96");        \
  GLB128(kb[2].u[0], (KADDR), "2048");      \
  GLB128(kb[2].u[1], (KADDR), "2080");      \
  GLB128(kb[3].u[0], (KADDR), "2112");      \
  GLB128(kb[3].u[1], (KADDR), "2144");

__global__ __launch_bounds__(128)
void flash_attn_kernel(const float* __restrict__ x,
                       const __bf16* __restrict__ Q,
                       const __bf16* __restrict__ K,
                       const __bf16* __restrict__ V,
                       const float* __restrict__ gamma,
                       float* __restrict__ outp)
{
  __shared__ __bf16 Plds[4][16 * 32];        // per-wave P tile (bf16)
  __shared__ float  Olds[4][128 * 20];       // per-wave epilogue transpose

  const int lane = threadIdx.x & 31;
  const int wave = threadIdx.x >> 5;
  const int half = lane >> 4;
  const int l15  = lane & 15;

  const int b  = blockIdx.x >> 8;
  const int i0 = (blockIdx.x & 255) * 16;
  const int c0 = wave * 128;

  const __bf16* Qb = Q + (size_t)b * 4096 * 64;
  const __bf16* Kb = K + (size_t)b * 4096 * 64;
  const __bf16* Vb = V + (size_t)b * 512 * 4096;

  Op16 qa0, qa1;
  Op16 kb[4];
  Op16 vb[8];

  // loop-invariant Q A-operands (asm so loadcnt bookkeeping stays exact)
  const char* qaddr = (const char*)(Qb + (size_t)(i0 + l15) * 64 + half * 8);
  GLB128(qa0.u[0], qaddr, "0");
  GLB128(qa0.u[1], qaddr, "32");
  GLB128(qa1.u[0], qaddr, "64");
  GLB128(qa1.u[1], qaddr, "96");

  Op16 ones;
#pragma unroll
  for (int i = 0; i < 16; ++i) ones.e[i] = 0x3F80;  // bf16(1.0)

  const char* kbase = (const char*)(Kb + (size_t)l15 * 64 + half * 8);          // + j*128B
  const char* vbase = (const char*)(Vb + (size_t)(c0 + l15) * 4096 + half * 8); // + j*2B

  float m = -3.0e38f;
  v8f accL = {};
  v8f acc[8] = {};
  unsigned tokf = 0;

  K2_LOAD_K(kbase);                          // K tiles for j0 = 0

  auto kvstep = [&](int j0, bool last) {
    // ---- wait for this step's K (token ordered into kb) ----
    {
      unsigned _t;
      WAIT_TOK("0x0", _t);
      kb[0].u[0].x |= _t; kb[1].u[0].x |= _t;
      kb[2].u[0].x |= _t; kb[3].u[0].x |= _t;
    }

    // ---- S = Q @ K^T (4 WMMAs) ----
    v8f s[2];
    {
      v8f z0 = {};
      z0 = __builtin_amdgcn_wmma_f32_16x16x32_bf16(false, qa0.v, false, kb[0].v, (short)0, z0, false, false);
      z0 = __builtin_amdgcn_wmma_f32_16x16x32_bf16(false, qa1.v, false, kb[1].v, (short)0, z0, false, false);
      s[0] = z0;
      v8f z1 = {};
      z1 = __builtin_amdgcn_wmma_f32_16x16x32_bf16(false, qa0.v, false, kb[2].v, (short)0, z1, false, false);
      z1 = __builtin_amdgcn_wmma_f32_16x16x32_bf16(false, qa1.v, false, kb[3].v, (short)0, z1, false, false);
      s[1] = z1;
    }

    // ---- issue V loads: 16 x b128, one base + tile offsets (t*16*4096*2) ----
    const char* va = vbase + (size_t)j0 * 2;
    GLB128(vb[0].u[0], va, "0");       GLB128(vb[0].u[1], va, "32");
    GLB128(vb[1].u[0], va, "131072");  GLB128(vb[1].u[1], va, "131104");
    GLB128(vb[2].u[0], va, "262144");  GLB128(vb[2].u[1], va, "262176");
    GLB128(vb[3].u[0], va, "393216");  GLB128(vb[3].u[1], va, "393248");
    GLB128(vb[4].u[0], va, "524288");  GLB128(vb[4].u[1], va, "524320");
    GLB128(vb[5].u[0], va, "655360");  GLB128(vb[5].u[1], va, "655392");
    GLB128(vb[6].u[0], va, "786432");  GLB128(vb[6].u[1], va, "786464");
    GLB128(vb[7].u[0], va, "917504");  GLB128(vb[7].u[1], va, "917536");

    // ---- issue next-step K loads (skipped on last iteration) ----
    if (!last) {
      const char* ka = kbase + (size_t)(j0 + 32) * 128;
      K2_LOAD_K(ka);
    }

    // pin softmax after the load-issue point (token into s)
    {
      unsigned _p;
      TOK_ONLY(_p);
      s[0][0] = __uint_as_float(__float_as_uint(s[0][0]) | _p);
    }

    // ---- tile-wide running max (wave-uniform); covers V/K load latency ----
    float tm = -3.0e38f;
#pragma unroll
    for (int v = 0; v < 8; ++v) tm = fmaxf(tm, fmaxf(s[0][v], s[1][v]));
#pragma unroll
    for (int msk = 1; msk < 32; msk <<= 1) tm = fmaxf(tm, __shfl_xor(tm, msk, 32));

    float mnew = fmaxf(m, tm);
    float scl  = __expf(m - mnew);
    m = mnew;
    if (scl < 1.0f) {                        // uniform branch; rare after warmup
#pragma unroll
      for (int t = 0; t < 8; ++t)
#pragma unroll
        for (int v = 0; v < 8; ++v) acc[t][v] *= scl;
#pragma unroll
      for (int v = 0; v < 8; ++v) accL[v] *= scl;
    }

    // ---- P = exp(S - m), restripe through wave-private LDS ----
#pragma unroll
    for (int v = 0; v < 8; ++v) {
      float p0 = __expf(s[0][v] - mnew);
      float p1 = __expf(s[1][v] - mnew);
      int r = v + half * 8;
      Plds[wave][r * 32 + l15]      = (__bf16)p0;
      Plds[wave][r * 32 + 16 + l15] = (__bf16)p1;
    }
    Op16 pa;
    {
      const __bf16* pr = &Plds[wave][l15 * 32 + half * 8];
      pa.u[0] = *(const uint4*)(pr);
      pa.u[1] = *(const uint4*)(pr + 16);
    }

    // ---- denominator: accL += P @ ones ----
    accL = __builtin_amdgcn_wmma_f32_16x16x32_bf16(
        false, pa.v, false, ones.v, (short)0, accL, false, false);

    // ---- wait for V (token ordered into vb) ----
    {
      unsigned _t;
      if (!last) { WAIT_TOK("0x8", _t); }    // next-K (8) still in flight
      else       { WAIT_TOK("0x0", _t); }    // drain everything
      vb[0].u[0].x |= _t; vb[1].u[0].x |= _t; vb[2].u[0].x |= _t; vb[3].u[0].x |= _t;
      vb[4].u[0].x |= _t; vb[5].u[0].x |= _t; vb[6].u[0].x |= _t; vb[7].u[0].x |= _t;
      tokf = _t;
    }

    // ---- O += P @ V (8 channel tiles) ----
#pragma unroll
    for (int t = 0; t < 8; ++t) {
      acc[t] = __builtin_amdgcn_wmma_f32_16x16x32_bf16(
          false, pa.v, false, vb[t].v, (short)0, acc[t], false, false);
    }
  };

  for (int j0 = 0; j0 < 4064; j0 += 32) kvstep(j0, false);
  kvstep(4064, true);                        // peeled: no next-K, full drain

  // ---- epilogue (token-poisoned): normalize, LDS transpose, coalesced I/O ----
#pragma unroll
  for (int v = 0; v < 8; ++v)
    accL[v] = __uint_as_float(__float_as_uint(accL[v]) | tokf);

  float*       ol   = &Olds[wave][0] + tokf;
  const float* xe   = x + tokf;
  float*       ope  = outp + tokf;
  const int    i0e  = i0 + (int)tokf;
  const int    c0e  = c0 + (int)tokf;
  const int    l15e = l15 + (int)tokf;
  const int    hfe  = half + (int)tokf;
  const int    lne  = lane + (int)tokf;

#pragma unroll
  for (int v = 0; v < 8; ++v) {
    float inv = 1.0f / accL[v];
    int il = v + hfe * 8;
#pragma unroll
    for (int t = 0; t < 8; ++t) {
      ol[(t * 16 + l15e) * 20 + il] = acc[t][v] * inv;   // [c_local][i_local]
    }
  }
  const float g = gamma[0];
#pragma unroll
  for (int cc = 0; cc < 4; ++cc) {
    int cl = lne * 4 + cc;                               // 0..127
    const float* orow = ol + cl * 20;
    size_t base = ((size_t)b * 512 + c0e + cl) * 4096 + i0e;
#pragma unroll
    for (int q4 = 0; q4 < 4; ++q4) {
      float4 o  = *(const float4*)(orow + q4 * 4);
      float4 xr = *(const float4*)(xe + base + q4 * 4);
      float4 rr;
      rr.x = g * o.x + xr.x;
      rr.y = g * o.y + xr.y;
      rr.z = g * o.z + xr.z;
      rr.w = g * o.w + xr.w;
      *(float4*)(ope + base + q4 * 4) = rr;
    }
  }
}

// ---------------------------------------------------------------------------
extern "C" void kernel_launch(void* const* d_in, const int* in_sizes, int n_in,
                              void* d_out, int out_size, void* d_ws, size_t ws_size,
                              hipStream_t stream) {
  const float* x     = (const float*)d_in[0];
  // d_in[1] = s (unused by reference)
  const float* wq    = (const float*)d_in[2];
  const float* bq    = (const float*)d_in[3];
  const float* wk    = (const float*)d_in[4];
  const float* bk    = (const float*)d_in[5];
  const float* wv    = (const float*)d_in[6];
  const float* bv    = (const float*)d_in[7];
  const float* gamma = (const float*)d_in[8];
  float* outp = (float*)d_out;

  __bf16* ws  = (__bf16*)d_ws;
  __bf16* Xt  = ws;                                   // [2,4096,512]
  __bf16* Q   = Xt  + (size_t)2 * 4096 * 512;         // [2,4096,64]
  __bf16* K   = Q   + (size_t)2 * 4096 * 64;          // [2,4096,64]
  __bf16* V   = K   + (size_t)2 * 4096 * 64;          // [2,512,4096]
  __bf16* Wqb = V   + (size_t)2 * 512 * 4096;         // [64,512]
  __bf16* Wkb = Wqb + (size_t)64 * 512;
  __bf16* Wvb = Wkb + (size_t)64 * 512;               // [512,512]

  xpose_kernel<<<dim3(2 * 512), dim3(256), 0, stream>>>(x, Xt);
  cvt_kernel<<<dim3(128),  dim3(256), 0, stream>>>(wq, Wqb, 64 * 512);
  cvt_kernel<<<dim3(128),  dim3(256), 0, stream>>>(wk, Wkb, 64 * 512);
  cvt_kernel<<<dim3(1024), dim3(256), 0, stream>>>(wv, Wvb, 512 * 512);

  qkv_project_kernel<<<dim3((2 * 10 * 256) / 4), dim3(128), 0, stream>>>(
      Xt, Wqb, bq, Wkb, bk, Wvb, bv, Q, K, V);

  flash_attn_kernel<<<dim3(2 * 256), dim3(128), 0, stream>>>(
      x, Q, K, V, gamma, outp);
}
